// DGCNN_partseg_5806795784776
// MI455X (gfx1250) — matrix operand order repaired and checked by reading, not compiled, not run on hardware
//
#include <hip/hip_runtime.h>
#include <hip/hip_bf16.h>

#define NEG_SLOPE 0.2f
#define CDIV(a, b) (((a) + (b) - 1) / (b))

typedef __attribute__((ext_vector_type(16))) __bf16 v16bf;
typedef __attribute__((ext_vector_type(8)))  float  v8f;

union ABFrag { v16bf v; unsigned short s[16]; };

__device__ __forceinline__ unsigned short f2bf(float f) {
  union { float f; unsigned u; } x; x.f = f;
  unsigned r = x.u + 0x7FFFu + ((x.u >> 16) & 1u);
  return (unsigned short)(r >> 16);
}
__device__ __forceinline__ float lrelu(float y) { return y > 0.f ? y : NEG_SLOPE * y; }

// A-matrix (16x32 bf16) lane layout: element i of v16bf for lane-half lhi -> K offset
// (one contiguous 16-element run per lane: k = lhi*16 + i, matching B staging)
__device__ __forceinline__ int a_k_of(int i, int lhi) {
  return lhi * 16 + i;
}

// ---------------------------------------------------------------------------
// KNN: thread per (b,n); iterative min with (dist, idx) lexicographic keys.
// Matches top_k(-d^2): ascending distance, ties broken by lower index.
// ---------------------------------------------------------------------------
template <int C, int K>
__global__ void knn_kernel(const float* __restrict__ feat, int* __restrict__ idxout,
                           int B_, int Npts) {
  int t = blockIdx.x * blockDim.x + threadIdx.x;
  if (t >= B_ * Npts) return;
  int b = t / Npts, n = t % Npts;
  const float* fb = feat + (long)b * C * Npts;
  float nn = 0.f;
  for (int c = 0; c < C; ++c) { float v = fb[c * Npts + n]; nn += v * v; }
  float lastD = -__builtin_inff(); int lastI = -1;
  for (int j = 0; j < K; ++j) {
    float bD = __builtin_inff(); int bI = 0;
    for (int m = 0; m < Npts; ++m) {
      float dot = 0.f, mm = 0.f;
      for (int c = 0; c < C; ++c) {
        float fm = fb[c * Npts + m];
        float fn = fb[c * Npts + n];
        dot += fn * fm; mm += fm * fm;
      }
      float d = nn + mm - 2.f * dot;
      bool valid  = (d > lastD) || (d == lastD && m > lastI);
      bool better = (d < bD)   || (d == bD && m < bI);
      if (valid && better) { bD = d; bI = m; }
    }
    idxout[(long)(b * Npts + n) * K + j] = bI;
    lastD = bD; lastI = bI;
  }
}

// ---------------------------------------------------------------------------
// Fused EdgeConv: build edge rows [xj-xi, xi] in LDS (bf16), WMMA GEMM(s)
// with fused BN+LeakyReLU, then max over k neighbors.
// Weights are staged in LDS pre-swizzled into B-fragment order:
//   sW[((kc*4 + ct)*32 + lane)*16 + i] = W[o = ct*16+(lane&15)][k = kc*32+(lane>>4)*16+i]
// so each B fragment is one 32-byte contiguous per-lane LDS read.
// One WG handles TP points -> R = TP*KNB = 80 rows (5 row tiles x 4 col tiles).
// ---------------------------------------------------------------------------
template <int CIN, int KNB, int TP, bool TWO_CONV>
__global__ __launch_bounds__(256) void edgeconv_kernel(
    const float* __restrict__ feat, const int* __restrict__ idx,
    const float* __restrict__ w1, const float* __restrict__ g1, const float* __restrict__ b1,
    const float* __restrict__ w2, const float* __restrict__ g2, const float* __restrict__ b2,
    float* __restrict__ out, int Npts) {
  constexpr int C    = CIN / 2;
  constexpr int CINP = (CIN <= 32) ? 32 : 128;
  constexpr int R    = TP * KNB;       // 80
  constexpr int RT   = R / 16;         // 5
  constexpr int NJOB = RT * 4;         // 20

  __shared__ unsigned short sW1[CINP * 64];          // fragment-swizzled
  __shared__ unsigned short sW2[TWO_CONV ? 64 * 64 : 64];
  __shared__ unsigned short sMid[TWO_CONV ? R * 64 : 64];
  __shared__ float sRes[R * 64];       // 20KB; edge buffer aliases this
  unsigned short* sEdge = (unsigned short*)sRes;

  int tid  = threadIdx.x;
  int nblk = Npts / TP;
  int b    = blockIdx.x / nblk;
  int p0   = (blockIdx.x % nblk) * TP;

  // stage W1 swizzled: t = ((kc*4 + ct)*32 + lane)*16 + i
  for (int t = tid; t < CINP * 64; t += 256) {
    int i = t & 15, ln = (t >> 4) & 31, ctk = t >> 9;
    int ct = ctk & 3, kc = ctk >> 2;
    int k = kc * 32 + (ln >> 4) * 16 + i;
    int o = ct * 16 + (ln & 15);
    sW1[t] = (k < CIN) ? f2bf(w1[o * CIN + k]) : (unsigned short)0;
  }
  if (TWO_CONV) {
    for (int t = tid; t < 64 * 64; t += 256) {
      int i = t & 15, ln = (t >> 4) & 31, ctk = t >> 9;
      int ct = ctk & 3, kc = ctk >> 2;
      int k = kc * 32 + (ln >> 4) * 16 + i;
      int o = ct * 16 + (ln & 15);
      sW2[t] = f2bf(w2[o * 64 + k]);
    }
  }
  // build edge rows (row-major, A fragments read as 2x b128 per lane)
  for (int t = tid; t < R * CINP; t += 256) {
    int r = t / CINP, c = t % CINP;
    int p = r / KNB, j = r % KNB;
    int n = p0 + p;
    float v = 0.f;
    if (c < CIN) {
      int nb = idx[((long)b * Npts + n) * KNB + j];
      if (c < C) v = feat[((long)b * C + c) * Npts + nb] - feat[((long)b * C + c) * Npts + n];
      else       v = feat[((long)b * C + (c - C)) * Npts + n];
    }
    sEdge[t] = f2bf(v);
  }
  __syncthreads();

  int lane = tid & 31, wv = tid >> 5;
  int lhi = lane >> 4, lm = lane & 15;
  float accs[3][8];

  // ---- stage 1 GEMM: out1 = LReLU(BN1(W1 * edge)) ----
#pragma unroll
  for (int jj = 0; jj < 3; ++jj) {
    int job = wv + jj * 8;
    if (job < NJOB) {
      int rt = job >> 2, ct = job & 3;
      v8f acc = {};
      for (int kc = 0; kc < CINP / 32; ++kc) {
        ABFrag af, bf2;
        const unsigned short* ap = &sEdge[(rt * 16 + lm) * CINP + kc * 32];
        const unsigned short* wp = &sW1[(((kc * 4 + ct) * 32) + lane) * 16];
#pragma unroll
        for (int i = 0; i < 16; ++i) {
          af.s[i] = ap[a_k_of(i, lhi)];
          bf2.s[i] = wp[i];
        }
        acc = __builtin_amdgcn_wmma_f32_16x16x32_bf16(false, af.v, false, bf2.v,
                                                      (short)0, acc, false, false);
      }
      int o = ct * 16 + lm;
      float gg = g1[o], bb = b1[o];
#pragma unroll
      for (int r8 = 0; r8 < 8; ++r8) {
        float y = lrelu(acc[r8] * gg + bb);
        if (TWO_CONV) {
          int row = rt * 16 + r8 + lhi * 8;
          sMid[row * 64 + o] = f2bf(y);
        } else {
          accs[jj][r8] = y;
        }
      }
    }
  }
  __syncthreads();

  // ---- stage 2 GEMM (optional): out2 = LReLU(BN2(W2 * out1)) ----
  if (TWO_CONV) {
#pragma unroll
    for (int jj = 0; jj < 3; ++jj) {
      int job = wv + jj * 8;
      if (job < NJOB) {
        int rt = job >> 2, ct = job & 3;
        v8f acc = {};
        for (int kc = 0; kc < 2; ++kc) {
          ABFrag af, bf2;
          const unsigned short* ap = &sMid[(rt * 16 + lm) * 64 + kc * 32];
          const unsigned short* wp = &sW2[(((kc * 4 + ct) * 32) + lane) * 16];
#pragma unroll
          for (int i = 0; i < 16; ++i) {
            af.s[i] = ap[a_k_of(i, lhi)];
            bf2.s[i] = wp[i];
          }
          acc = __builtin_amdgcn_wmma_f32_16x16x32_bf16(false, af.v, false, bf2.v,
                                                        (short)0, acc, false, false);
        }
        int o = ct * 16 + lm;
        float gg = g2[o], bb = b2[o];
#pragma unroll
        for (int r8 = 0; r8 < 8; ++r8) accs[jj][r8] = lrelu(acc[r8] * gg + bb);
      }
    }
    __syncthreads();
  }

  // ---- scatter tile results to LDS ----
#pragma unroll
  for (int jj = 0; jj < 3; ++jj) {
    int job = wv + jj * 8;
    if (job < NJOB) {
      int rt = job >> 2, ct = job & 3;
      int o = ct * 16 + lm;
#pragma unroll
      for (int r8 = 0; r8 < 8; ++r8) {
        int row = rt * 16 + r8 + lhi * 8;
        sRes[row * 64 + o] = accs[jj][r8];
      }
    }
  }
  __syncthreads();

  // ---- max over neighbors ----
  for (int t = tid; t < TP * 64; t += 256) {
    int p = t / 64, o = t % 64;
    float m = -__builtin_inff();
    for (int j = 0; j < KNB; ++j) m = fmaxf(m, sRes[(p * KNB + j) * 64 + o]);
    out[((long)b * 64 + o) * Npts + p0 + p] = m;
  }
}

// ---------------------------------------------------------------------------
// Pooling: score -> rank-count top-k select -> gather+scale
// ---------------------------------------------------------------------------
__global__ void score_kernel(const float* __restrict__ feat, const float* __restrict__ pw,
                             const float* __restrict__ pb, float* __restrict__ sc,
                             int B_, int Npts) {
  int t = blockIdx.x * blockDim.x + threadIdx.x;
  if (t >= B_ * Npts) return;
  int b = t / Npts, n = t % Npts;
  float s = pb[0];
  for (int c = 0; c < 64; ++c) s += pw[c] * feat[((long)b * 64 + c) * Npts + n];
  sc[t] = 1.f / (1.f + expf(-s));
}

__global__ void rank_select_kernel(const float* __restrict__ sc, int* __restrict__ selidx,
                                   float* __restrict__ selval, int B_, int Npts, int npts) {
  int t = blockIdx.x * blockDim.x + threadIdx.x;
  if (t >= B_ * Npts) return;
  int b = t / Npts, n = t % Npts;
  float s = sc[b * Npts + n];
  int rank = 0;
  for (int m = 0; m < Npts; ++m) {
    float sm = sc[b * Npts + m];
    rank += (sm > s) || (sm == s && m < n);
  }
  if (rank < npts) { selidx[b * npts + rank] = n; selval[b * npts + rank] = s; }
}

__global__ void pool_gather_kernel(const float* __restrict__ xyz, const float* __restrict__ feat,
                                   const int* __restrict__ selidx, const float* __restrict__ selval,
                                   float* __restrict__ node, float* __restrict__ featsel,
                                   float* __restrict__ xyzsel, int B_, int Npts, int npts) {
  int t = blockIdx.x * blockDim.x + threadIdx.x;
  if (t >= B_ * npts) return;
  int b = t / npts, j = t % npts;
  int n = selidx[b * npts + j];
  float v = selval[b * npts + j];
  for (int c = 0; c < 3; ++c) {
    float x = xyz[((long)b * 3 + c) * Npts + n];
    xyzsel[((long)b * 3 + c) * npts + j] = x;
    node[((long)b * 3 + c) * npts + j] = x * v;
  }
  for (int c = 0; c < 64; ++c)
    featsel[((long)b * 64 + c) * npts + j] = feat[((long)b * 64 + c) * Npts + n] * v;
}

// ---------------------------------------------------------------------------
// Global max over last axis (for 256-dim global feature)
// ---------------------------------------------------------------------------
__global__ void maxlast_kernel(const float* __restrict__ in, float* __restrict__ out,
                               int B_, int C, int Npts, int outStride, int base) {
  int t = blockIdx.x * blockDim.x + threadIdx.x;
  if (t >= B_ * C) return;
  int b = t / C, c = t % C;
  float m = -__builtin_inff();
  for (int n = 0; n < Npts; ++n) m = fmaxf(m, in[((long)b * C + c) * Npts + n]);
  out[b * outStride + base + c] = m;
}

// ---------------------------------------------------------------------------
// Tiny MLP on (B, K) vectors: y = LReLU(g*(W x)+b), optional 2-src concat
// ---------------------------------------------------------------------------
__global__ void mlp_kernel(const float* __restrict__ xa, int Ca, const float* __restrict__ xb,
                           int Cb, const float* __restrict__ W, const float* __restrict__ g,
                           const float* __restrict__ bb, float* __restrict__ out, int B_, int O) {
  int t = blockIdx.x * blockDim.x + threadIdx.x;
  if (t >= B_ * O) return;
  int b = t / O, o = t % O;
  int K = Ca + Cb;
  float s = 0.f;
  for (int c = 0; c < K; ++c) {
    float x = (c < Ca) ? xa[b * Ca + c] : xb[b * Cb + (c - Ca)];
    s += W[o * K + c] * x;
  }
  out[b * O + o] = lrelu(s * g[o] + bb[o]);
}

// ---------------------------------------------------------------------------
// Stage concat([a, b]) channels into bf16 row-major rows for WMMA GEMM
// a: (B,Ca,Npts) or broadcast (B,Ca) when abc=1; b: (B,Cb,Npts)
// ---------------------------------------------------------------------------
__global__ void concat_rows_kernel(const float* __restrict__ a, int Ca, int abc,
                                   const float* __restrict__ bsrc, int Cb,
                                   unsigned short* __restrict__ xrow, int B_, int Npts) {
  int K = Ca + Cb;
  long t = (long)blockIdx.x * blockDim.x + threadIdx.x;
  long tot = (long)B_ * Npts * K;
  if (t >= tot) return;
  int c = (int)(t % K);
  long row = t / K;
  int b = (int)(row / Npts), n = (int)(row % Npts);
  float v;
  if (c < Ca) v = abc ? a[b * Ca + c] : a[((long)b * Ca + c) * Npts + n];
  else        v = bsrc[((long)b * Cb + (c - Ca)) * Npts + n];
  xrow[t] = f2bf(v);
}

// ---------------------------------------------------------------------------
// WMMA GEMM over row buffer: out(b,o,n) = [LReLU(g*.+b)]( W(O,K) . xrow )
// 256-thread block = 8 waves handle 8 row tiles of one 16-wide column tile;
// the weight panel is staged once in LDS, bf16, B-fragment swizzled.
// grid.x = Rtot/128 (Rtot multiple of 128), grid.y = ceil(O/16).
// ---------------------------------------------------------------------------
template <int K>
__global__ __launch_bounds__(256) void gemm_rows_kernel(
    const unsigned short* __restrict__ xrow, const float* __restrict__ W,
    const float* __restrict__ g, const float* __restrict__ bb,
    float* __restrict__ out, int O, int Npts, int doAct) {
  __shared__ unsigned short sW[(K / 32) * 32 * 16];
  int ct = blockIdx.y;
  int tid = threadIdx.x;
  for (int t = tid; t < (K / 32) * 512; t += 256) {
    int i = t & 15, ln = (t >> 4) & 31, kc = t >> 9;
    int k = kc * 32 + (ln >> 4) * 16 + i;
    int o = ct * 16 + (ln & 15);
    sW[t] = (o < O) ? f2bf(W[(long)o * K + k]) : (unsigned short)0;
  }
  __syncthreads();

  int lane = tid & 31, wv = tid >> 5;
  int lhi = lane >> 4, lm = lane & 15;
  int rt = blockIdx.x * 8 + wv;
  v8f acc = {};
  for (int kc = 0; kc < K / 32; ++kc) {
    ABFrag af, bf2;
    const unsigned short* ap = &xrow[(long)(rt * 16 + lm) * K + kc * 32];
    const unsigned short* wp = &sW[(kc * 32 + lane) * 16];
#pragma unroll
    for (int i = 0; i < 16; ++i) {
      af.s[i] = ap[a_k_of(i, lhi)];
      bf2.s[i] = wp[i];
    }
    acc = __builtin_amdgcn_wmma_f32_16x16x32_bf16(false, af.v, false, bf2.v,
                                                  (short)0, acc, false, false);
  }
  int o = ct * 16 + lm;
  if (o < O) {
    float gg = doAct ? g[o] : 1.f;
    float bo = doAct ? bb[o] : 0.f;
#pragma unroll
    for (int r8 = 0; r8 < 8; ++r8) {
      int row = rt * 16 + r8 + lhi * 8;
      float y = acc[r8];
      if (doAct) y = lrelu(y * gg + bo);
      int b = row / Npts, n = row % Npts;
      out[((long)b * O + o) * Npts + n] = y;
    }
  }
}

// ---------------------------------------------------------------------------
// Unpool: 3-NN indices + softmax over raw distances (faithful), then apply
// ---------------------------------------------------------------------------
__global__ void unpool_nn_kernel(const float* __restrict__ src, int M,
                                 const float* __restrict__ unk, int Nu,
                                 int* __restrict__ ui, float* __restrict__ uw, int B_) {
  int t = blockIdx.x * blockDim.x + threadIdx.x;
  if (t >= B_ * Nu) return;
  int b = t / Nu, n = t % Nu;
  float ux = unk[((long)b * 3 + 0) * Nu + n];
  float uy = unk[((long)b * 3 + 1) * Nu + n];
  float uz = unk[((long)b * 3 + 2) * Nu + n];
  float bd0 = __builtin_inff(), bd1 = bd0, bd2 = bd0;
  int bi0 = 0, bi1 = 0, bi2 = 0;
  for (int m = 0; m < M; ++m) {
    float dx = ux - src[((long)b * 3 + 0) * M + m];
    float dy = uy - src[((long)b * 3 + 1) * M + m];
    float dz = uz - src[((long)b * 3 + 2) * M + m];
    float d = dx * dx + dy * dy + dz * dz;
    if (d < bd0)      { bd2 = bd1; bi2 = bi1; bd1 = bd0; bi1 = bi0; bd0 = d; bi0 = m; }
    else if (d < bd1) { bd2 = bd1; bi2 = bi1; bd1 = d; bi1 = m; }
    else if (d < bd2) { bd2 = d; bi2 = m; }
  }
  // softmax over raw distances (matches reference's softmax(-neg_dist))
  float mx = fmaxf(bd0, fmaxf(bd1, bd2));
  float e0 = expf(bd0 - mx), e1 = expf(bd1 - mx), e2 = expf(bd2 - mx);
  float inv = 1.f / (e0 + e1 + e2);
  long base = (long)(b * Nu + n) * 3;
  ui[base + 0] = bi0; ui[base + 1] = bi1; ui[base + 2] = bi2;
  uw[base + 0] = e0 * inv; uw[base + 1] = e1 * inv; uw[base + 2] = e2 * inv;
}

__global__ void unpool_apply_kernel(const float* __restrict__ feat, const int* __restrict__ ui,
                                    const float* __restrict__ uw, float* __restrict__ out,
                                    int B_, int C, int Nu, int M) {
  long t = (long)blockIdx.x * blockDim.x + threadIdx.x;
  long tot = (long)B_ * C * Nu;
  if (t >= tot) return;
  int b = (int)(t / ((long)C * Nu));
  int rem = (int)(t % ((long)C * Nu));
  int c = rem / Nu, n = rem % Nu;
  long base = (long)(b * Nu + n) * 3;
  float r = 0.f;
#pragma unroll
  for (int i = 0; i < 3; ++i)
    r += uw[base + i] * feat[((long)b * C + c) * M + ui[base + i]];
  out[t] = r;
}

// ---------------------------------------------------------------------------
extern "C" void kernel_launch(void* const* d_in, const int* in_sizes, int n_in,
                              void* d_out, int out_size, void* d_ws, size_t ws_size,
                              hipStream_t stream) {
  (void)in_sizes; (void)n_in; (void)out_size; (void)ws_size;
  const int B = 8, N = 2048, N1 = 512, N2 = 128, N3 = 32;

  const float* x = (const float*)d_in[0];
  const float* l = (const float*)d_in[1];
  int pi = 2;
  const float* w1  = (const float*)d_in[pi++];
  const float* w2  = (const float*)d_in[pi++];
  const float* w3  = (const float*)d_in[pi++];
  const float* w4  = (const float*)d_in[pi++];
  const float* w5  = (const float*)d_in[pi++];
  const float* w6  = (const float*)d_in[pi++];
  const float* w6m = (const float*)d_in[pi++];
  const float* w7  = (const float*)d_in[pi++];
  const float* w8  = (const float*)d_in[pi++];
  const float* w9  = (const float*)d_in[pi++];
  const float* w10 = (const float*)d_in[pi++];
  const float* w11 = (const float*)d_in[pi++];
  const float* w12 = (const float*)d_in[pi++];
  const float* w13 = (const float*)d_in[pi++];
  const float* p1w = (const float*)d_in[pi++];
  const float* p2w = (const float*)d_in[pi++];
  const float* p3w = (const float*)d_in[pi++];
  const float* g1  = (const float*)d_in[pi++];  const float* b1  = (const float*)d_in[pi++];
  const float* g2  = (const float*)d_in[pi++];  const float* b2  = (const float*)d_in[pi++];
  const float* g3  = (const float*)d_in[pi++];  const float* b3  = (const float*)d_in[pi++];
  const float* g4  = (const float*)d_in[pi++];  const float* b4  = (const float*)d_in[pi++];
  const float* g5  = (const float*)d_in[pi++];  const float* b5  = (const float*)d_in[pi++];
  const float* g6  = (const float*)d_in[pi++];  const float* b6  = (const float*)d_in[pi++];
  const float* g6m = (const float*)d_in[pi++];  const float* b6m = (const float*)d_in[pi++];
  const float* g7  = (const float*)d_in[pi++];  const float* b7  = (const float*)d_in[pi++];
  const float* g8  = (const float*)d_in[pi++];  const float* b8  = (const float*)d_in[pi++];
  const float* g9  = (const float*)d_in[pi++];  const float* b9  = (const float*)d_in[pi++];
  const float* g10 = (const float*)d_in[pi++];  const float* b10 = (const float*)d_in[pi++];
  const float* g11 = (const float*)d_in[pi++];  const float* b11 = (const float*)d_in[pi++];
  const float* g12 = (const float*)d_in[pi++];  const float* b12 = (const float*)d_in[pi++];
  const float* p1b = (const float*)d_in[pi++];
  const float* p2b = (const float*)d_in[pi++];
  const float* p3b = (const float*)d_in[pi++];

  // outputs: seg, node1, node2, node3, n1s, n2s
  float* seg   = (float*)d_out;
  float* node1 = seg + 8 * 50 * 2048;
  float* node2 = node1 + 8 * 3 * 512;
  float* node3 = node2 + 8 * 3 * 128;
  float* n1s   = node3 + 8 * 3 * 32;
  float* n2s   = n1s + 8 * 3 * 512;

  // workspace carve-out
  char* wsp = (char*)d_ws;
  auto carve = [&](size_t bytes) -> void* {
    void* p = (void*)wsp;
    wsp += (bytes + 255) & ~(size_t)255;
    return p;
  };
  int*   idx1 = (int*)  carve((size_t)B * N * 40 * 4);
  int*   idx2 = (int*)  carve((size_t)B * N1 * 20 * 4);
  int*   idx3 = (int*)  carve((size_t)B * N2 * 10 * 4);
  int*   idx4 = (int*)  carve((size_t)B * N3 * 5 * 4);
  float* x1   = (float*)carve((size_t)B * 64 * N * 4);
  float* x2   = (float*)carve((size_t)B * 64 * N1 * 4);
  float* x3   = (float*)carve((size_t)B * 64 * N2 * 4);
  float* x4   = (float*)carve((size_t)B * 64 * N3 * 4);
  float* sc   = (float*)carve((size_t)B * N * 4);
  int*   seli = (int*)  carve((size_t)B * N1 * 4);
  float* selv = (float*)carve((size_t)B * N1 * 4);
  float* nf1  = (float*)carve((size_t)B * 64 * N1 * 4);
  float* nf2  = (float*)carve((size_t)B * 64 * N2 * 4);
  float* nf3  = (float*)carve((size_t)B * 64 * N3 * 4);
  float* n3s  = (float*)carve((size_t)B * 3 * N3 * 4);
  float* xt   = (float*)carve((size_t)B * 256 * 4);
  float* h6m  = (float*)carve((size_t)B * 1024 * 4);
  float* lf   = (float*)carve((size_t)B * 64 * 4);
  float* h8   = (float*)carve((size_t)B * 256 * 4);
  unsigned short* xrow = (unsigned short*)carve((size_t)B * N * 320 * 2);
  float* h9   = (float*)carve((size_t)B * 256 * N3 * 4);
  float* up1  = (float*)carve((size_t)B * 256 * N2 * 4);
  float* h10  = (float*)carve((size_t)B * 256 * N2 * 4);
  float* up2  = (float*)carve((size_t)B * 256 * N1 * 4);
  float* h11  = (float*)carve((size_t)B * 256 * N1 * 4);
  float* up3  = (float*)carve((size_t)B * 256 * N * 4);
  float* h12  = (float*)carve((size_t)B * 128 * N * 4);
  int*   ui   = (int*)  carve((size_t)B * N * 3 * 4);
  float* uw   = (float*)carve((size_t)B * N * 3 * 4);

  // ---------------- Level 1 ----------------
  knn_kernel<3, 40><<<CDIV(B * N, 256), 256, 0, stream>>>(x, idx1, B, N);
  edgeconv_kernel<6, 40, 2, true><<<B * (N / 2), 256, 0, stream>>>(
      x, idx1, w1, g1, b1, w2, g2, b2, x1, N);
  score_kernel<<<CDIV(B * N, 256), 256, 0, stream>>>(x1, p1w, p1b, sc, B, N);
  rank_select_kernel<<<CDIV(B * N, 256), 256, 0, stream>>>(sc, seli, selv, B, N, N1);
  pool_gather_kernel<<<CDIV(B * N1, 256), 256, 0, stream>>>(
      x, x1, seli, selv, node1, nf1, n1s, B, N, N1);

  // ---------------- Level 2 ----------------
  knn_kernel<64, 20><<<CDIV(B * N1, 256), 256, 0, stream>>>(nf1, idx2, B, N1);
  edgeconv_kernel<128, 20, 4, true><<<B * (N1 / 4), 256, 0, stream>>>(
      nf1, idx2, w3, g3, b3, w4, g4, b4, x2, N1);
  score_kernel<<<CDIV(B * N1, 256), 256, 0, stream>>>(x2, p2w, p2b, sc, B, N1);
  rank_select_kernel<<<CDIV(B * N1, 256), 256, 0, stream>>>(sc, seli, selv, B, N1, N2);
  pool_gather_kernel<<<CDIV(B * N2, 256), 256, 0, stream>>>(
      n1s, x2, seli, selv, node2, nf2, n2s, B, N1, N2);

  // ---------------- Level 3 ----------------
  knn_kernel<64, 10><<<CDIV(B * N2, 256), 256, 0, stream>>>(nf2, idx3, B, N2);
  edgeconv_kernel<128, 10, 8, false><<<B * (N2 / 8), 256, 0, stream>>>(
      nf2, idx3, w5, g5, b5, nullptr, nullptr, nullptr, x3, N2);
  score_kernel<<<CDIV(B * N2, 256), 256, 0, stream>>>(x3, p3w, p3b, sc, B, N2);
  rank_select_kernel<<<CDIV(B * N2, 256), 256, 0, stream>>>(sc, seli, selv, B, N2, N3);
  pool_gather_kernel<<<CDIV(B * N3, 256), 256, 0, stream>>>(
      n2s, x3, seli, selv, node3, nf3, n3s, B, N2, N3);

  // ---------------- Level 4 ----------------
  knn_kernel<64, 5><<<CDIV(B * N3, 256), 256, 0, stream>>>(nf3, idx4, B, N3);
  edgeconv_kernel<128, 5, 16, false><<<B * (N3 / 16), 256, 0, stream>>>(
      nf3, idx4, w6, g6, b6, nullptr, nullptr, nullptr, x4, N3);

  // ---------------- Global feature ----------------
  maxlast_kernel<<<CDIV(B * 64, 256), 256, 0, stream>>>(x1, xt, B, 64, N, 256, 0);
  maxlast_kernel<<<CDIV(B * 64, 256), 256, 0, stream>>>(x2, xt, B, 64, N1, 256, 64);
  maxlast_kernel<<<CDIV(B * 64, 256), 256, 0, stream>>>(x3, xt, B, 64, N2, 256, 128);
  maxlast_kernel<<<CDIV(B * 64, 256), 256, 0, stream>>>(x4, xt, B, 64, N3, 256, 192);
  mlp_kernel<<<CDIV(B * 1024, 256), 256, 0, stream>>>(xt, 256, nullptr, 0, w6m, g6m, b6m, h6m, B, 1024);
  mlp_kernel<<<CDIV(B * 64, 256), 256, 0, stream>>>(l, 16, nullptr, 0, w7, g7, b7, lf, B, 64);
  mlp_kernel<<<CDIV(B * 256, 256), 256, 0, stream>>>(h6m, 1024, lf, 64, w8, g8, b8, h8, B, 256);

  // ---------------- Decoder ----------------
  // w9: concat(broadcast h8 [256], x4 [64]) over N3
  concat_rows_kernel<<<CDIV(B * N3 * 320, 256), 256, 0, stream>>>(h8, 256, 1, x4, 64, xrow, B, N3);
  gemm_rows_kernel<320><<<dim3(B * N3 / 128, 16), 256, 0, stream>>>(xrow, w9, g9, b9, h9, 256, N3, 1);

  unpool_nn_kernel<<<CDIV(B * N2, 256), 256, 0, stream>>>(n3s, N3, n2s, N2, ui, uw, B);
  unpool_apply_kernel<<<CDIV(B * 256 * N2, 256), 256, 0, stream>>>(h9, ui, uw, up1, B, 256, N2, N3);

  concat_rows_kernel<<<CDIV(B * N2 * 320, 256), 256, 0, stream>>>(up1, 256, 0, x3, 64, xrow, B, N2);
  gemm_rows_kernel<320><<<dim3(B * N2 / 128, 16), 256, 0, stream>>>(xrow, w10, g10, b10, h10, 256, N2, 1);

  unpool_nn_kernel<<<CDIV(B * N1, 256), 256, 0, stream>>>(n2s, N2, n1s, N1, ui, uw, B);
  unpool_apply_kernel<<<CDIV(B * 256 * N1, 256), 256, 0, stream>>>(h10, ui, uw, up2, B, 256, N1, N2);

  concat_rows_kernel<<<CDIV(B * N1 * 320, 256), 256, 0, stream>>>(up2, 256, 0, x2, 64, xrow, B, N1);
  gemm_rows_kernel<320><<<dim3(B * N1 / 128, 16), 256, 0, stream>>>(xrow, w11, g11, b11, h11, 256, N1, 1);

  unpool_nn_kernel<<<CDIV(B * N, 256), 256, 0, stream>>>(n1s, N1, x, N, ui, uw, B);
  unpool_apply_kernel<<<CDIV(B * 256 * N, 256), 256, 0, stream>>>(h11, ui, uw, up3, B, 256, N, N1);

  concat_rows_kernel<<<CDIV(B * N * 320, 256), 256, 0, stream>>>(up3, 256, 0, x1, 64, xrow, B, N);
  gemm_rows_kernel<320><<<dim3(B * N / 128, 8), 256, 0, stream>>>(xrow, w12, g12, b12, h12, 128, N, 1);

  // seg head: plain einsum (no BN / activation)
  concat_rows_kernel<<<CDIV(B * N * 128, 256), 256, 0, stream>>>(h12, 128, 0, nullptr, 0, xrow, B, N);
  gemm_rows_kernel<128><<<dim3(B * N / 128, 4), 256, 0, stream>>>(xrow, w13, nullptr, nullptr, seg, 50, N, 0);
}
